// GlobalMatch_77824807403843
// MI455X (gfx1250) — matrix operand order repaired and checked by reading, not compiled, not run on hardware
//
#include <hip/hip_runtime.h>
#include <math.h>

typedef __attribute__((ext_vector_type(16))) __bf16 v16bf;
typedef __attribute__((ext_vector_type(8)))  float  v8f;
typedef __attribute__((ext_vector_type(4)))  unsigned int u32x4;

union Frag {            // 32 bytes = 16 bf16 = one WMMA A/B operand (8 VGPRs)
    u32x4 q[2];
    v16bf v;
};

#define EPS 1e-6f
#define NC 256          // channels (K of GEMM)
#define NP 4096         // patches per batch (64x64)
#define NB 4

// ---------------------------------------------------------------------------
// Kernel 1: gather strided patches -> contiguous bf16 hi/lo rows [b][n][c],
// fused per-row reduction: isB==0 -> s[n] = a2 + 2*eps*sa + C*eps^2
//                          isB==1 -> t[m] = b2 - 2*eps*sb
// grid: NB*NP blocks, 256 threads (one thread per channel c)
// ---------------------------------------------------------------------------
__global__ __launch_bounds__(256) void pack_reduce_kernel(
    const float* __restrict__ x, unsigned short* __restrict__ hi,
    unsigned short* __restrict__ lo, float* __restrict__ red, int isB)
{
    const int g  = blockIdx.x;
    const int b  = g >> 12;
    const int n  = g & (NP - 1);
    const int ph = n >> 6;
    const int pw = n & 63;
    const int c  = threadIdx.x;

    const float v = x[(((size_t)b * NC + c) * 256 + 4 * ph) * 256 + 4 * pw];

    // split v = hi + lo (both bf16, RNE)
    unsigned u  = __float_as_uint(v);
    unsigned hb = (u + 0x7FFFu + ((u >> 16) & 1u)) >> 16;
    float    hf = __uint_as_float(hb << 16);
    float    rr = v - hf;
    unsigned u2 = __float_as_uint(rr);
    unsigned lb = (u2 + 0x7FFFu + ((u2 >> 16) & 1u)) >> 16;

    const size_t oidx = ((size_t)b * NP + n) * NC + c;   // coalesced along c
    hi[oidx] = (unsigned short)hb;
    lo[oidx] = (unsigned short)lb;

    // block reduction of sum(v) and sum(v*v)
    float s1 = v, s2 = v * v;
    for (int off = 16; off >= 1; off >>= 1) {
        s1 += __shfl_xor(s1, off, 32);
        s2 += __shfl_xor(s2, off, 32);
    }
    __shared__ float r1[8], r2[8];
    if ((threadIdx.x & 31) == 0) {
        r1[threadIdx.x >> 5] = s1;
        r2[threadIdx.x >> 5] = s2;
    }
    __syncthreads();
    if (threadIdx.x == 0) {
        float t1 = 0.f, t2 = 0.f;
        for (int w = 0; w < 8; ++w) { t1 += r1[w]; t2 += r2[w]; }
        float res = isB ? (t2 - 2.0f * EPS * t1)
                        : (t2 + 2.0f * EPS * t1 + (float)NC * EPS * EPS);
        red[(size_t)b * NP + n] = res;
    }
}

// ---------------------------------------------------------------------------
// Kernel 2: fused GEMM (bf16 hi/lo split, fp32 accumulate via WMMA) + column
// min + sqrt/sigmoid + 4x4 nearest upsample.
// grid: 128 WGs x 256 threads; 8 waves/WG, one 16-row A tile per wave.
// ---------------------------------------------------------------------------
__global__ __launch_bounds__(256) void gemm_min_kernel(
    const unsigned short* __restrict__ ahi, const unsigned short* __restrict__ alo,
    const unsigned short* __restrict__ bhi, const unsigned short* __restrict__ blo,
    const float* __restrict__ sA, const float* __restrict__ tB,
    float* __restrict__ out)
{
    const int lane  = threadIdx.x & 31;
    const int wave  = threadIdx.x >> 5;
    const int batch = blockIdx.x >> 5;            // 32 WGs per batch
    const int tileBase = (blockIdx.x & 31) * 8;   // 8 n-tiles per WG
    const int n0   = (tileBase + wave) * 16;
    const int l16  = lane & 15;
    const int half = lane >> 4;

    // ---- preload this wave's A fragments (all K, hi+lo) into registers ----
    // A 16x32 bf16 layout: lanes 0-15 hold K {k0..k0+7, k0+16..k0+23} of row M=lane,
    //                      lanes 16-31 hold K {k0+8..k0+15, k0+24..k0+31}.
    const size_t arow = ((size_t)batch * NP + n0 + l16) * NC;   // element idx
    Frag Ahi[8], Alo[8];
#pragma unroll
    for (int k = 0; k < 8; ++k) {
        const int k0 = 32 * k;
        const unsigned short* pa = ahi + arow + k0 + half * 8;
        const unsigned short* pl = alo + arow + k0 + half * 8;
        Ahi[k].q[0] = *(const u32x4*)(pa);
        Ahi[k].q[1] = *(const u32x4*)(pa + 16);
        Alo[k].q[0] = *(const u32x4*)(pl);
        Alo[k].q[1] = *(const u32x4*)(pl + 16);
    }

    float minv[8];
#pragma unroll
    for (int r = 0; r < 8; ++r) minv[r] = 3.0e38f;

    const size_t bb = (size_t)batch * NP;

    // ---- stream all 256 B tiles, fused min epilogue ----
    for (int mt = 0; mt < 256; ++mt) {
        const int m0 = mt * 16;
        // B 32x16 bf16 layout: lanes 0-15: K {k0..k0+7, k0+16..k0+23} of column N=lane,
        //                      lanes 16-31: K {k0+8..k0+15, k0+24..k0+31}.
        const size_t brow = (bb + m0 + l16) * NC;
        v8f acc = {};
#pragma unroll
        for (int k = 0; k < 8; ++k) {
            const int koff = 32 * k + half * 8;
            Frag Bh, Bl;
            const unsigned short* pb = bhi + brow + koff;
            const unsigned short* pl = blo + brow + koff;
            Bh.q[0] = *(const u32x4*)(pb);
            Bh.q[1] = *(const u32x4*)(pb + 16);
            Bl.q[0] = *(const u32x4*)(pl);
            Bl.q[1] = *(const u32x4*)(pl + 16);
            // ab ~= ahi*bhi + ahi*blo + alo*bhi  (fp32 accumulate)
            acc = __builtin_amdgcn_wmma_f32_16x16x32_bf16(
                false, Ahi[k].v, false, Bh.v, (short)0, acc, false, false);
            acc = __builtin_amdgcn_wmma_f32_16x16x32_bf16(
                false, Ahi[k].v, false, Bl.v, (short)0, acc, false, false);
            acc = __builtin_amdgcn_wmma_f32_16x16x32_bf16(
                false, Alo[k].v, false, Bh.v, (short)0, acc, false, false);
        }
        // D layout: VGPR r, lanes 0-15: (row r, col lane); lanes 16-31: (row 8+r, col lane-16)
        const float tcol = tB[bb + m0 + l16];
#pragma unroll
        for (int r = 0; r < 8; ++r)
            minv[r] = fminf(minv[r], tcol - 2.0f * acc[r]);
    }

    // ---- min across the 16 columns held by each 16-lane half ----
#pragma unroll
    for (int r = 0; r < 8; ++r) {
        float v = minv[r];
        for (int off = 8; off >= 1; off >>= 1)
            v = fminf(v, __shfl_xor(v, off, 32));
        minv[r] = v;
    }

    // lane 0 -> rows n0+0..7, lane 16 -> rows n0+8..15
    __shared__ float sc[8 * 16];
    if (l16 == 0) {
#pragma unroll
        for (int r = 0; r < 8; ++r) {
            const int rloc = r + half * 8;
            float d2 = sA[bb + n0 + rloc] + minv[r];
            d2 = fmaxf(d2, 0.0f);
            float l2 = sqrtf(d2);
            sc[wave * 16 + rloc] = 2.0f / (1.0f + expf(l2));
        }
    }
    __syncthreads();

    // ---- 4x4 nearest upsample: WG covers 128 consecutive n -> 2048 pixels ----
    const int nbase = tileBase * 16;
    for (int idx = threadIdx.x; idx < 128 * 16; idx += 256) {
        const int ln  = idx >> 4;
        const int sub = idx & 15;
        const int i   = sub >> 2;
        const int j   = sub & 3;
        const int n   = nbase + ln;
        const int ph  = n >> 6;
        const int pw  = n & 63;
        out[(size_t)batch * 65536 + (4 * ph + i) * 256 + 4 * pw + j] = sc[ln];
    }
}

// ---------------------------------------------------------------------------
extern "C" void kernel_launch(void* const* d_in, const int* in_sizes, int n_in,
                              void* d_out, int out_size, void* d_ws, size_t ws_size,
                              hipStream_t stream)
{
    const float* a = (const float*)d_in[0];
    const float* b = (const float*)d_in[1];
    float* out = (float*)d_out;

    const size_t nElems = (size_t)NB * NP * NC;   // 4*4096*256
    unsigned short* ahi = (unsigned short*)d_ws;
    unsigned short* alo = ahi + nElems;
    unsigned short* bhi = alo + nElems;
    unsigned short* blo = bhi + nElems;
    float* sA = (float*)(blo + nElems);
    float* tB = sA + (size_t)NB * NP;

    pack_reduce_kernel<<<NB * NP, 256, 0, stream>>>(a, ahi, alo, sA, 0);
    pack_reduce_kernel<<<NB * NP, 256, 0, stream>>>(b, bhi, blo, tB, 1);
    gemm_min_kernel<<<128, 256, 0, stream>>>(ahi, alo, bhi, blo, sA, tB, out);
}
